// FreqFusionBlock_39384850104702
// MI455X (gfx1250) — compile-verified, matrix operand order checked by reading, not run on hardware
//
#include <hip/hip_runtime.h>
#include <hip/hip_bf16.h>

typedef __bf16 bf16;
typedef __attribute__((ext_vector_type(16))) __bf16 v16bf;
typedef __attribute__((ext_vector_type(8)))  __bf16 v8bf;
typedef __attribute__((ext_vector_type(8)))  float  v8f;

#define HWDIM 56
#define NPIX  3136
#define PDIM  58            // padded spatial dim
#define PPIX  3364          // 58*58
#define CCH   256
#define NHEAD 8
#define HD    32

// ---------- scalar conversion helpers ----------
__device__ __forceinline__ bf16 f2bf(float f) {
  unsigned u = __builtin_bit_cast(unsigned, f);
  unsigned r = (u + 0x7FFFu + ((u >> 16) & 1u)) >> 16;
  unsigned short h = (unsigned short)r;
  return __builtin_bit_cast(bf16, h);
}

// Fragment slice: 2 contiguous 16B loads -> 16 bf16 (CDNA5 16x16x32 layouts).
// Caller passes p = base + rowsel*ld + kb  (rowsel = lane&15, kb = (lane>>4)*8).
__device__ __forceinline__ v16bf frag_ld(const bf16* p) {
  v8bf lo = *(const v8bf*)(p);
  v8bf hi = *(const v8bf*)(p + 16);
  return __builtin_shufflevector(lo, hi, 0, 1, 2, 3, 4, 5, 6, 7,
                                 8, 9, 10, 11, 12, 13, 14, 15);
}

#define WMMA_BF16(A, B, C) \
  __builtin_amdgcn_wmma_f32_16x16x32_bf16(false, (A), false, (B), (short)0, (C), false, false)

// =====================================================================
// Dense GEMM on transposed activations:
//   acc[M,N] = W[M,K] * X_T[N,K]^T + bias
// Wave computes a (16*MT) x 64 tile: MT A-frags x 4 B-frags -> 4*MT WMMAs
// per 32-wide K slice; every operand access is a b128 load.
// =====================================================================
template <int MT>
__global__ void gemm_wmma(const bf16* __restrict__ W, const bf16* __restrict__ X,
                          const float* __restrict__ bias,
                          float* __restrict__ outF_nm, float* __restrict__ outFT,
                          bf16* __restrict__ outBT, bf16* __restrict__ outB_nm,
                          int M, int Mreal, int N, int Kd, int act, float oscale) {
  int n0 = blockIdx.x * 64;
  int m0 = blockIdx.y * (16 * MT);
  int l = threadIdx.x & 31; int nl = l & 15; int kb = (l >> 4) * 8;
  v8f acc[MT][4] = {};
  for (int k0 = 0; k0 < Kd; k0 += 32) {
    v16bf a[MT];
#pragma unroll
    for (int t = 0; t < MT; ++t)
      a[t] = frag_ld(W + (size_t)(m0 + t * 16 + nl) * Kd + k0 + kb);
#pragma unroll
    for (int j = 0; j < 4; ++j) {
      const bf16* bp = X + (size_t)(n0 + j * 16 + nl) * Kd + k0 + kb;
      __builtin_prefetch(bp + 32, 0, 1);   // speculative, branch-free
      v16bf b = frag_ld(bp);
#pragma unroll
      for (int t = 0; t < MT; ++t) acc[t][j] = WMMA_BF16(a[t], b, acc[t][j]);
    }
  }
  int mb = (l >> 4) * 8;
#pragma unroll
  for (int t = 0; t < MT; ++t) {
    int mt0 = m0 + t * 16;
#pragma unroll
    for (int j = 0; j < 4; ++j) {
      int col = n0 + j * 16 + nl;
      float vr[8];
#pragma unroll
      for (int r = 0; r < 8; ++r) {
        int row = mt0 + mb + r;
        float v = acc[t][j][r] + ((row < Mreal) ? bias[row] : 0.0f);
        if (act == 1) v = (v >= 0.f) ? v : 0.2f * v;
        vr[r] = v * oscale;
      }
      if (outF_nm) {
#pragma unroll
        for (int r = 0; r < 8; ++r)
          if (mt0 + mb + r < Mreal) outF_nm[(size_t)(mt0 + mb + r) * N + col] = vr[r];
      }
      if (outB_nm) {
#pragma unroll
        for (int r = 0; r < 8; ++r)
          if (mt0 + mb + r < Mreal) outB_nm[(size_t)(mt0 + mb + r) * N + col] = f2bf(vr[r]);
      }
      if (outFT) {
        if (mt0 + 16 <= Mreal) {
          v8f pv;
#pragma unroll
          for (int r = 0; r < 8; ++r) pv[r] = vr[r];
          *(v8f*)(outFT + (size_t)col * M + mt0 + mb) = pv;
        } else {
#pragma unroll
          for (int r = 0; r < 8; ++r)
            if (mt0 + mb + r < Mreal) outFT[(size_t)col * M + mt0 + mb + r] = vr[r];
        }
      }
      if (outBT) {
        if (mt0 + 16 <= Mreal) {
          v8bf pk;
#pragma unroll
          for (int r = 0; r < 8; ++r) pk[r] = f2bf(vr[r]);
          *(v8bf*)(outBT + (size_t)col * M + mt0 + mb) = pk;
        } else {
#pragma unroll
          for (int r = 0; r < 8; ++r)
            if (mt0 + mb + r < Mreal) outBT[(size_t)col * M + mt0 + mb + r] = f2bf(vr[r]);
        }
      }
    }
  }
}

// =====================================================================
// 3x3 conv as implicit GEMM on ZERO-PADDED transposed activations:
//   Out[M,N] = Wp[M,9*Cin] * im2col(Xp) + bias
// Wp[o][t*Cin + c];  Xp: (58*58, Cin) bf16 with zero ring.  Tap loop is
// unrolled (compile-time dy/dx); inner K-slice loop is branch-free b128
// loads + 8 WMMAs -- identical shape to the dense GEMM loop.
// =====================================================================
__global__ void conv3_wmma(const bf16* __restrict__ Wp, const bf16* __restrict__ Xp,
                           const float* __restrict__ bias,
                           float* __restrict__ outF_nm, bf16* __restrict__ outBT,
                           int M, int Cin, int act) {
  const int N = NPIX;
  int n0 = blockIdx.x * 64;
  int m0 = blockIdx.y * 32;
  int l = threadIdx.x & 31; int nl = l & 15; int kb = (l >> 4) * 8;
  int Kd = 9 * Cin;
  size_t off[4];
#pragma unroll
  for (int j = 0; j < 4; ++j) {          // per-lane pixel, computed once
    int n = n0 + j * 16 + nl;
    int y = n / HWDIM, x = n % HWDIM;
    off[j] = (size_t)((y + 1) * PDIM + (x + 1)) * Cin + kb;
  }
  const bf16* aw0 = Wp + (size_t)(m0 + nl) * Kd + kb;
  const bf16* aw1 = Wp + (size_t)(m0 + 16 + nl) * Kd + kb;
  v8f acc[2][4] = {};
#pragma unroll
  for (int t = 0; t < 9; ++t) {
    const int shift = ((t / 3 - 1) * PDIM + (t % 3 - 1)) * CCH; // Cin==CCH here
    for (int c0 = 0; c0 < Cin; c0 += 32) {
      int k0 = t * Cin + c0;
      v16bf a0 = frag_ld(aw0 + k0);
      v16bf a1 = frag_ld(aw1 + k0);
#pragma unroll
      for (int j = 0; j < 4; ++j) {
        v16bf b = frag_ld(Xp + off[j] + shift + c0);
        acc[0][j] = WMMA_BF16(a0, b, acc[0][j]);
        acc[1][j] = WMMA_BF16(a1, b, acc[1][j]);
      }
    }
  }
  int mb = (l >> 4) * 8;
#pragma unroll
  for (int t = 0; t < 2; ++t) {
    int mt0 = m0 + t * 16;
#pragma unroll
    for (int j = 0; j < 4; ++j) {
      int col = n0 + j * 16 + nl;
      float vr[8];
#pragma unroll
      for (int r = 0; r < 8; ++r) {
        float v = acc[t][j][r] + bias[mt0 + mb + r];
        if (act == 1) v = (v >= 0.f) ? v : 0.2f * v;
        vr[r] = v;
      }
      if (outF_nm) {
#pragma unroll
        for (int r = 0; r < 8; ++r)
          outF_nm[(size_t)(mt0 + mb + r) * N + col] = vr[r];
      }
      if (outBT) {
        v8bf pk;
#pragma unroll
        for (int r = 0; r < 8; ++r) pk[r] = f2bf(vr[r]);
        *(v8bf*)(outBT + (size_t)col * M + mt0 + mb) = pk;
      }
    }
  }
}

// =====================================================================
// Flash attention, one wave = 16 query rows of one head (hd=32 == WMMA K).
// =====================================================================
__global__ void attn_wmma(const bf16* __restrict__ qT, const bf16* __restrict__ kvT,
                          const bf16* __restrict__ Vcn, bf16* __restrict__ aoT) {
  const int N = NPIX;
  int h  = blockIdx.y;
  int n0 = blockIdx.x * 16;
  __shared__ bf16 ldsP[16 * 32];

  int l = threadIdx.x & 31; int nl = l & 15; int kb = (l >> 4) * 8;
  int mb = kb;

  v16bf qa = frag_ld(qT + (size_t)(n0 + nl) * CCH + h * HD + kb);
  const bf16* Vh = Vcn + (size_t)(h * HD) * N;
  v8f o0 = {}, o1 = {};
  float mrun[8], lrun[8];
#pragma unroll
  for (int r = 0; r < 8; ++r) { mrun[r] = -1e30f; lrun[r] = 0.f; }

  for (int m0 = 0; m0 < N; m0 += 32) {
    v16bf kf0 = frag_ld(kvT + (size_t)(m0 + nl) * 512 + h * HD + kb);
    v16bf kf1 = frag_ld(kvT + (size_t)(m0 + 16 + nl) * 512 + h * HD + kb);
    v8f s0 = {}, s1 = {};
    s0 = WMMA_BF16(qa, kf0, s0);
    s1 = WMMA_BF16(qa, kf1, s1);
#pragma unroll
    for (int r = 0; r < 8; ++r) {
      float mx = fmaxf(s0[r], s1[r]);
#pragma unroll
      for (int d = 1; d < 16; d <<= 1) mx = fmaxf(mx, __shfl_xor(mx, d, 32));
      float mnew = fmaxf(mrun[r], mx);
      float p0 = __expf(s0[r] - mnew);
      float p1 = __expf(s1[r] - mnew);
      float corr = __expf(mrun[r] - mnew);
      float ps = p0 + p1;
#pragma unroll
      for (int d = 1; d < 16; d <<= 1) ps += __shfl_xor(ps, d, 32);
      lrun[r] = lrun[r] * corr + ps;
      mrun[r] = mnew;
      o0[r] *= corr; o1[r] *= corr;
      int row = mb + r;
      ldsP[row * 32 + nl]      = f2bf(p0);
      ldsP[row * 32 + 16 + nl] = f2bf(p1);
    }
    __syncthreads();
    v16bf pa  = frag_ld(ldsP + nl * 32 + kb);
    v16bf vb0 = frag_ld(Vh + (size_t)nl * N + m0 + kb);
    v16bf vb1 = frag_ld(Vh + (size_t)(16 + nl) * N + m0 + kb);
    o0 = WMMA_BF16(pa, vb0, o0);
    o1 = WMMA_BF16(pa, vb1, o1);
    __syncthreads();
  }
#pragma unroll
  for (int r = 0; r < 8; ++r) {
    float inv = 1.0f / lrun[r];
    size_t qrow = (size_t)(n0 + mb + r) * CCH + h * HD;
    aoT[qrow + nl]      = f2bf(o0[r] * inv);
    aoT[qrow + 16 + nl] = f2bf(o1[r] * inv);
  }
}

// =====================================================================
// elementwise / packing kernels
// =====================================================================
// (C,N) f32 -> (N,C) bf16 transpose-convert
__global__ void k_t_f32_to_bf16(const float* __restrict__ in, bf16* __restrict__ out,
                                int Crows) {
  int i = blockIdx.x * blockDim.x + threadIdx.x;
  if (i >= Crows * NPIX) return;
  int c = i % Crows, n = i / Crows;
  out[i] = f2bf(in[(size_t)c * NPIX + n]);
}
// (C,N) f32 -> (58*58, C) bf16, zero ring
__global__ void k_padT_bf16(const float* __restrict__ in, bf16* __restrict__ out) {
  int i = blockIdx.x * blockDim.x + threadIdx.x;
  if (i >= PPIX * CCH) return;
  int c = i % CCH, p = i / CCH;
  int yp = p / PDIM, xp = p % PDIM;
  bool in_b = (yp >= 1) & (yp <= HWDIM) & (xp >= 1) & (xp <= HWDIM);
  out[i] = in_b ? f2bf(in[(size_t)c * NPIX + (yp - 1) * HWDIM + (xp - 1)]) : f2bf(0.f);
}
// (C,N) f32 -> (C, 58*58) f32, zero ring
__global__ void k_pad_f32(const float* __restrict__ in, float* __restrict__ out) {
  int i = blockIdx.x * blockDim.x + threadIdx.x;
  if (i >= CCH * PPIX) return;
  int p = i % PPIX, c = i / PPIX;
  int yp = p / PDIM, xp = p % PDIM;
  bool in_b = (yp >= 1) & (yp <= HWDIM) & (xp >= 1) & (xp <= HWDIM);
  out[i] = in_b ? in[(size_t)c * NPIX + (yp - 1) * HWDIM + (xp - 1)] : 0.0f;
}
// a,b pixel-major (N,C) f32 -> (58*58, C) bf16 sum with zero ring
__global__ void k_add2_padT(const float* __restrict__ a, const float* __restrict__ b,
                            bf16* __restrict__ out) {
  int i = blockIdx.x * blockDim.x + threadIdx.x;
  if (i >= PPIX * CCH) return;
  int c = i % CCH, p = i / CCH;
  int yp = p / PDIM, xp = p % PDIM;
  bool in_b = (yp >= 1) & (yp <= HWDIM) & (xp >= 1) & (xp <= HWDIM);
  if (in_b) {
    size_t n = (size_t)((yp - 1) * HWDIM + (xp - 1)) * CCH + c;
    out[i] = f2bf(a[n] + b[n]);
  } else {
    out[i] = f2bf(0.f);
  }
}
__global__ void k_pack_mat(const float* __restrict__ w, bf16* __restrict__ out, int n) {
  int i = blockIdx.x * blockDim.x + threadIdx.x;
  if (i < n) out[i] = f2bf(w[i]);
}
__global__ void k_pack_pad16(const float* __restrict__ w, bf16* __restrict__ out,
                             int rowsReal, int cols) {
  int i = blockIdx.x * blockDim.x + threadIdx.x;
  int total = 16 * cols;
  if (i < total) {
    int r = i / cols, c = i % cols;
    out[i] = (r < rowsReal) ? f2bf(w[r * cols + c]) : f2bf(0.0f);
  }
}
__global__ void k_pack_conv(const float* __restrict__ w, bf16* __restrict__ out,
                            int O, int Cin) {
  int i = blockIdx.x * blockDim.x + threadIdx.x;
  int total = O * Cin * 9;
  if (i < total) {
    int o = i / (Cin * 9);
    int c = (i / 9) % Cin;
    int t = i % 9;
    out[(size_t)o * 9 * Cin + t * Cin + c] = f2bf(w[i]);
  }
}
// depthwise 3x3 from padded (C,58,58) f32 -> (N,C) bf16 ; branch-free taps
__global__ void k_dw3x3(const float* __restrict__ xp, const float* __restrict__ w,
                        const float* __restrict__ b, bf16* __restrict__ out) {
  int i = blockIdx.x * blockDim.x + threadIdx.x;
  if (i >= CCH * NPIX) return;
  int c = i % CCH, n = i / CCH;
  int y = n / HWDIM, x = n % HWDIM;
  const float* src = xp + (size_t)c * PPIX + y * PDIM + x;   // top-left tap
  float acc = b[c];
#pragma unroll
  for (int ty = 0; ty < 3; ++ty)
#pragma unroll
    for (int tx = 0; tx < 3; ++tx)
      acc += src[ty * PDIM + tx] * w[c * 9 + ty * 3 + tx];
  out[i] = f2bf(acc);
}
// softmax over 9 taps; buffer (N,16) pixel-major
__global__ void k_softmax9(float* __restrict__ buf) {
  int n = blockIdx.x * blockDim.x + threadIdx.x;
  if (n >= NPIX) return;
  float v[9], mx = -1e30f;
#pragma unroll
  for (int t = 0; t < 9; ++t) { v[t] = buf[n * 16 + t]; mx = fmaxf(mx, v[t]); }
  float s = 0.f;
#pragma unroll
  for (int t = 0; t < 9; ++t) { v[t] = __expf(v[t] - mx); s += v[t]; }
  float inv = 1.0f / s;
#pragma unroll
  for (int t = 0; t < 9; ++t) buf[n * 16 + t] = v[t] * inv;
}
// f_smooth_T[n,c] = sum_t dec_p[c, n+t] * kwT[n,t] ; branch-free taps
__global__ void k_apply_lpf(const float* __restrict__ decp, const float* __restrict__ kwT,
                            bf16* __restrict__ out) {
  int i = blockIdx.x * blockDim.x + threadIdx.x;
  if (i >= CCH * NPIX) return;
  int c = i % CCH, n = i / CCH;
  int y = n / HWDIM, x = n % HWDIM;
  const float* src = decp + (size_t)c * PPIX + y * PDIM + x;
  float s = 0.f;
#pragma unroll
  for (int t = 0; t < 9; ++t)
    s += src[(t / 3) * PDIM + (t % 3)] * kwT[n * 16 + t];
  out[i] = f2bf(s);
}
// f_high_T[n,c] = 2*enc - sum_t enc_p[c, n+t] * wlpT[n,t]
__global__ void k_apply_hpf(const float* __restrict__ encp, const float* __restrict__ wlpT,
                            float* __restrict__ out) {
  int i = blockIdx.x * blockDim.x + threadIdx.x;
  if (i >= CCH * NPIX) return;
  int c = i % CCH, n = i / CCH;
  int y = n / HWDIM, x = n % HWDIM;
  const float* src = encp + (size_t)c * PPIX + y * PDIM + x;
  float s = 2.0f * src[PDIM + 1];                 // center tap == enc value
#pragma unroll
  for (int t = 0; t < 9; ++t)
    s -= src[(t / 3) * PDIM + (t % 3)] * wlpT[n * 16 + t];
  out[i] = s;
}

// =====================================================================
extern "C" void kernel_launch(void* const* d_in, const int* in_sizes, int n_in,
                              void* d_out, int out_size, void* d_ws, size_t ws_size,
                              hipStream_t stream) {
  const float* dec       = (const float*)d_in[0];
  const float* enc       = (const float*)d_in[1];
  const float* alpf_dw_w = (const float*)d_in[2];
  const float* alpf_dw_b = (const float*)d_in[3];
  const float* alpf_p1_w = (const float*)d_in[4];
  const float* alpf_p1_b = (const float*)d_in[5];
  const float* alpf_p2_w = (const float*)d_in[6];
  const float* alpf_p2_b = (const float*)d_in[7];
  const float* ahpf_c1_w = (const float*)d_in[8];
  const float* ahpf_c1_b = (const float*)d_in[9];
  const float* ahpf_c2_w = (const float*)d_in[10];
  const float* ahpf_c2_b = (const float*)d_in[11];
  const float* q_w  = (const float*)d_in[12];
  const float* q_b  = (const float*)d_in[13];
  const float* kv_w = (const float*)d_in[14];
  const float* kv_b = (const float*)d_in[15];
  const float* out_w = (const float*)d_in[16];
  const float* out_b = (const float*)d_in[17];
  const float* fusion_w = (const float*)d_in[18];
  const float* fusion_b = (const float*)d_in[19];
  float* outp = (float*)d_out;

  const int N = NPIX;
  size_t off = 0;
  auto alloc = [&](size_t bytes) {
    void* p = (char*)d_ws + off;
    off += (bytes + 255) & ~(size_t)255;
    return p;
  };
  // bf16 activation buffers
  bf16* enc_T   = (bf16*)alloc((size_t)N * CCH * 2);        // (N,C)
  bf16* enc_Tp  = (bf16*)alloc((size_t)PPIX * CCH * 2);     // padded (58*58,C)
  bf16* dw_T    = (bf16*)alloc((size_t)N * CCH * 2);
  bf16* t128_T  = (bf16*)alloc((size_t)N * 128 * 2);
  bf16* fs_T    = (bf16*)alloc((size_t)N * CCH * 2);
  bf16* q_T     = (bf16*)alloc((size_t)N * CCH * 2);
  bf16* kv_T    = (bf16*)alloc((size_t)N * 512 * 2);
  bf16* kv_nm   = (bf16*)alloc((size_t)512 * N * 2);        // channel-major (V half)
  bf16* ao_T    = (bf16*)alloc((size_t)N * CCH * 2);
  bf16* h1_T    = (bf16*)alloc((size_t)N * 128 * 2);
  bf16* fsum_Tp = (bf16*)alloc((size_t)PPIX * CCH * 2);     // padded (58*58,C)
  // packed bf16 weights
  bf16* w_p1  = (bf16*)alloc((size_t)128 * 256 * 2);
  bf16* w_p2  = (bf16*)alloc((size_t)16 * 128 * 2);
  bf16* w_q   = (bf16*)alloc((size_t)256 * 256 * 2);
  bf16* w_kv  = (bf16*)alloc((size_t)512 * 256 * 2);
  bf16* w_out = (bf16*)alloc((size_t)256 * 256 * 2);
  bf16* w_ac2 = (bf16*)alloc((size_t)16 * 128 * 2);
  bf16* w_ac1 = (bf16*)alloc((size_t)128 * 2304 * 2);
  bf16* w_fus = (bf16*)alloc((size_t)256 * 2304 * 2);
  // f32 scratch
  float* dec_p      = (float*)alloc((size_t)CCH * PPIX * 4); // padded (C,58,58)
  float* enc_p      = (float*)alloc((size_t)CCH * PPIX * 4);
  float* kwT        = (float*)alloc((size_t)N * 16 * 4);
  float* wlpT       = (float*)alloc((size_t)N * 16 * 4);
  float* f_alignedT = (float*)alloc((size_t)N * CCH * 4);
  float* fhT        = (float*)alloc((size_t)N * CCH * 4);

  const int TB = 256;
  auto blk = [](int n, int t) { return (n + t - 1) / t; };
  const int CN = CCH * N;
  const int CP = CCH * PPIX;

  // ---- packing / padding ----
  k_pack_mat  <<<blk(128 * 256, TB), TB, 0, stream>>>(alpf_p1_w, w_p1, 128 * 256);
  k_pack_pad16<<<blk(16 * 128, TB), TB, 0, stream>>>(alpf_p2_w, w_p2, 9, 128);
  k_pack_mat  <<<blk(256 * 256, TB), TB, 0, stream>>>(q_w, w_q, 256 * 256);
  k_pack_mat  <<<blk(512 * 256, TB), TB, 0, stream>>>(kv_w, w_kv, 512 * 256);
  k_pack_mat  <<<blk(256 * 256, TB), TB, 0, stream>>>(out_w, w_out, 256 * 256);
  k_pack_pad16<<<blk(16 * 128, TB), TB, 0, stream>>>(ahpf_c2_w, w_ac2, 9, 128);
  k_pack_conv <<<blk(128 * 256 * 9, TB), TB, 0, stream>>>(ahpf_c1_w, w_ac1, 128, 256);
  k_pack_conv <<<blk(256 * 256 * 9, TB), TB, 0, stream>>>(fusion_w, w_fus, 256, 256);
  k_t_f32_to_bf16<<<blk(CN, TB), TB, 0, stream>>>(enc, enc_T, CCH);
  k_padT_bf16<<<blk(CP, TB), TB, 0, stream>>>(enc, enc_Tp);
  k_pad_f32  <<<blk(CP, TB), TB, 0, stream>>>(dec, dec_p);
  k_pad_f32  <<<blk(CP, TB), TB, 0, stream>>>(enc, enc_p);

  // ---- ALPF branch ----
  k_dw3x3<<<blk(CN, TB), TB, 0, stream>>>(dec_p, alpf_dw_w, alpf_dw_b, dw_T);
  gemm_wmma<2><<<dim3(N / 64, 128 / 32), 32, 0, stream>>>(w_p1, dw_T, alpf_p1_b,
            nullptr, nullptr, t128_T, nullptr, 128, 128, N, 256, /*lrelu*/1, 1.0f);
  gemm_wmma<1><<<dim3(N / 64, 1), 32, 0, stream>>>(w_p2, t128_T, alpf_p2_b,
            nullptr, kwT, nullptr, nullptr, 16, 9, N, 128, 0, 1.0f);
  k_softmax9<<<blk(N, TB), TB, 0, stream>>>(kwT);
  k_apply_lpf<<<blk(CN, TB), TB, 0, stream>>>(dec_p, kwT, fs_T);

  // ---- cross attention ----
  const float qscale = 0.17677669529663689f;   // 1/sqrt(hd=32), folded into q
  gemm_wmma<2><<<dim3(N / 64, 256 / 32), 32, 0, stream>>>(w_q, fs_T, q_b,
            nullptr, nullptr, q_T, nullptr, 256, 256, N, 256, 0, qscale);
  gemm_wmma<2><<<dim3(N / 64, 512 / 32), 32, 0, stream>>>(w_kv, enc_T, kv_b,
            nullptr, nullptr, kv_T, kv_nm, 512, 512, N, 256, 0, 1.0f);
  attn_wmma<<<dim3(N / 16, NHEAD), 32, 0, stream>>>(q_T, kv_T,
            kv_nm + (size_t)256 * N, ao_T);
  gemm_wmma<2><<<dim3(N / 64, 256 / 32), 32, 0, stream>>>(w_out, ao_T, out_b,
            nullptr, f_alignedT, nullptr, nullptr, 256, 256, N, 256, 0, 1.0f);

  // ---- AHPF branch ----
  conv3_wmma<<<dim3(N / 64, 128 / 32), 32, 0, stream>>>(w_ac1, enc_Tp, ahpf_c1_b,
            nullptr, h1_T, 128, 256, /*lrelu*/1);
  gemm_wmma<1><<<dim3(N / 64, 1), 32, 0, stream>>>(w_ac2, h1_T, ahpf_c2_b,
            nullptr, wlpT, nullptr, nullptr, 16, 9, N, 128, 0, 1.0f);
  k_softmax9<<<blk(N, TB), TB, 0, stream>>>(wlpT);
  k_apply_hpf<<<blk(CN, TB), TB, 0, stream>>>(enc_p, wlpT, fhT);

  // ---- fusion 3x3 conv ----
  k_add2_padT<<<blk(CP, TB), TB, 0, stream>>>(f_alignedT, fhT, fsum_Tp);
  conv3_wmma<<<dim3(N / 64, 256 / 32), 32, 0, stream>>>(w_fus, fsum_Tp, fusion_b,
            outp, nullptr, 256, 256, 0);
}
